// DeformConv_4389456576585
// MI455X (gfx1250) — matrix operand order repaired and checked by reading, not compiled, not run on hardware
//
#include <hip/hip_runtime.h>

typedef __attribute__((ext_vector_type(16))) __bf16    v16bf;
typedef __attribute__((ext_vector_type(8)))  float     v8f;
typedef __attribute__((ext_vector_type(4)))  unsigned  v4u;

#define CIN  256
#define COUT 256
#define HH   64
#define WW   64
#define KKK  9
#define HW   (HH * WW)
#define PITCH 72           // bf16 row pitch: 144B rows, 16B aligned
#define NWT  (KKK * 8 * COUT * 32)   // 589824 bf16 per weight plane

// Split f32 into (hi, lo) bf16 pair: hi = truncated top 16 bits, lo = RNE bf16 of remainder.
__device__ __forceinline__ void split_bf16(float v, unsigned short& h, unsigned short& l) {
  unsigned u = __float_as_uint(v);
  h = (unsigned short)(u >> 16);
  float r = v - __uint_as_float(u & 0xffff0000u);
  unsigned ur = __float_as_uint(r);
  ur += 0x7fffu + ((ur >> 16) & 1u);     // round-to-nearest-even to bf16
  l = (unsigned short)(ur >> 16);
}

// ---------------------------------------------------------------------------
// Kernel 1: split weight [Cout, C, 3, 3] into hi/lo bf16 planes, pre-swizzled
// to the 16-bit A 16x32 lane layout. Record per (kk, c32-block, o): 32 bf16 =
// [s=lane-half][16], where lane-half s=0 carries K{0-15}, s=1 K{16-31}.
// ---------------------------------------------------------------------------
__global__ __launch_bounds__(256) void pack_weight_kernel(
    const float* __restrict__ w, unsigned short* __restrict__ wtH,
    unsigned short* __restrict__ wtL) {
  int i = blockIdx.x * 256 + threadIdx.x;   // over NWT
  if (i >= NWT) return;
  int t  = i & 15;
  int s  = (i >> 4) & 1;
  int o  = (i >> 5) & 255;
  int cb = (i >> 13) & 7;                    // c32 block within tap
  int kk = i >> 16;
  int kl = 16 * s + t;  // K within 32-chunk
  int c  = cb * 32 + kl;
  float v = w[(size_t)o * (CIN * KKK) + c * KKK + kk];
  unsigned short h, l;
  split_bf16(v, h, l);
  wtH[i] = h;
  wtL[i] = l;
}

// ---------------------------------------------------------------------------
// Kernel 2: fused deformable-im2col + bf16x3 WMMA GEMM (f32 accumulate).
// Grid: 256 blocks = (b, h). Block: 512 threads = 16 waves.
// Wave w owns Cout tile [16w,16w+16) x 64 output columns (4 N-tiles).
// K = 2304 processed as 36 chunks of 64 (tap x 64 channels), double-buffered.
// ---------------------------------------------------------------------------
__global__ __launch_bounds__(512, 1) void deform_conv_wmma_kernel(
    const float* __restrict__ x, const float* __restrict__ offs,
    const unsigned short* __restrict__ wtH, const unsigned short* __restrict__ wtL,
    float* __restrict__ out) {

  __shared__ float          s_w[KKK][WW][4];     // bilinear weights (validity folded)
  __shared__ int            s_i[KKK][WW][4];     // clamped corner offsets in HW plane
  __shared__ unsigned short sH[2][64 * PITCH];   // cols tile, hi bf16 [pixel][K]
  __shared__ unsigned short sL[2][64 * PITCH];   // cols tile, lo bf16 [pixel][K]

  const int tid = threadIdx.x;
  const int b = blockIdx.x >> 6;
  const int h = blockIdx.x & 63;

  // --- Phase 0: sampling parameters for 9 taps x 64 output columns ---
  for (int t = tid; t < KKK * WW; t += 512) {
    int kk = t >> 6;
    int j  = t & 63;
    const float* ob = offs + (((size_t)(b * 2 * KKK + kk * 2) * HH + h) * WW + j);
    float dy = ob[0];
    float dx = ob[HW];
    float py = (float)(h - 1 + (kk / 3)) + dy;
    float px = (float)(j - 1 + (kk % 3)) + dx;
    float y0f = floorf(py), x0f = floorf(px);
    float fy = py - y0f, fx = px - x0f;
    int y0 = (int)y0f, x0 = (int)x0f;
    int y1 = y0 + 1,  x1 = x0 + 1;
    float vy0 = (y0 >= 0 && y0 < HH) ? 1.f : 0.f;
    float vy1 = (y1 >= 0 && y1 < HH) ? 1.f : 0.f;
    float vx0 = (x0 >= 0 && x0 < WW) ? 1.f : 0.f;
    float vx1 = (x1 >= 0 && x1 < WW) ? 1.f : 0.f;
    int cy0 = min(max(y0, 0), HH - 1), cy1 = min(max(y1, 0), HH - 1);
    int cx0 = min(max(x0, 0), WW - 1), cx1 = min(max(x1, 0), WW - 1);
    s_w[kk][j][0] = (1.f - fy) * (1.f - fx) * vy0 * vx0;
    s_w[kk][j][1] = (1.f - fy) * fx         * vy0 * vx1;
    s_w[kk][j][2] = fy         * (1.f - fx) * vy1 * vx0;
    s_w[kk][j][3] = fy         * fx         * vy1 * vx1;
    s_i[kk][j][0] = cy0 * WW + cx0;
    s_i[kk][j][1] = cy0 * WW + cx1;
    s_i[kk][j][2] = cy1 * WW + cx0;
    s_i[kk][j][3] = cy1 * WW + cx1;
  }
  __syncthreads();

  const int lane = tid & 31;
  const int wave = tid >> 5;
  const int lrow = lane & 15;
  const int lhi  = lane >> 4;
  const int o0   = wave << 4;

  v8f acc[4] = {};

  const int jj = tid & 63;   // build: output column sampled by this thread
  const int cg = tid >> 6;   // build: channel group (8 channels)

  // Gather-build one K-chunk (tap, 64 channels) into hi/lo bf16 tiles.
  auto build = [&](int q, int buf) {
    int kk = q >> 2;
    int c0 = (q & 3) << 6;
    float w0 = s_w[kk][jj][0], w1 = s_w[kk][jj][1];
    float w2 = s_w[kk][jj][2], w3 = s_w[kk][jj][3];
    int   i0 = s_i[kk][jj][0], i1 = s_i[kk][jj][1];
    int   i2 = s_i[kk][jj][2], i3 = s_i[kk][jj][3];
    const float* xb = x + (size_t)(b * CIN + c0 + cg * 8) * HW;
    unsigned short hs[8], ls[8];
#pragma unroll
    for (int cc = 0; cc < 8; ++cc) {
      const float* xp = xb + (size_t)cc * HW;
      float v = w0 * xp[i0] + w1 * xp[i1] + w2 * xp[i2] + w3 * xp[i3];
      split_bf16(v, hs[cc], ls[cc]);
    }
    v4u hv, lv;
#pragma unroll
    for (int k = 0; k < 4; ++k) {
      hv[k] = (unsigned)hs[2 * k] | ((unsigned)hs[2 * k + 1] << 16);
      lv[k] = (unsigned)ls[2 * k] | ((unsigned)ls[2 * k + 1] << 16);
    }
    int idx = jj * PITCH + cg * 8;          // 16B aligned (144B rows)
    *(v4u*)&sH[buf][idx] = hv;
    *(v4u*)&sL[buf][idx] = lv;
  };

  // Consume one K-chunk: 2 K=32 steps, each = A(hi,lo) load + 4 N-tiles x 3 WMMA.
  auto consume = [&](int q, int buf) {
    int kk  = q >> 2;
    int cb2 = (q & 3) << 1;
#pragma unroll
    for (int ks = 0; ks < 2; ++ks) {
      int cb = cb2 + ks;
      size_t aoff = (((size_t)((kk * 8 + cb) * COUT + o0 + lrow)) * 2 + lhi) * 16;
      v16bf Ah = *(const v16bf*)(wtH + aoff);
      v16bf Al = *(const v16bf*)(wtL + aoff);
#pragma unroll
      for (int nt = 0; nt < 4; ++nt) {
        int bidx = (nt * 16 + lrow) * PITCH + ks * 32 + lhi * 16;
        v16bf Bh = *(const v16bf*)&sH[buf][bidx];
        v16bf Bl = *(const v16bf*)&sL[buf][bidx];
        acc[nt] = __builtin_amdgcn_wmma_f32_16x16x32_bf16(
            false, Ah, false, Bl, (short)0, acc[nt], false, false);
        acc[nt] = __builtin_amdgcn_wmma_f32_16x16x32_bf16(
            false, Al, false, Bh, (short)0, acc[nt], false, false);
        acc[nt] = __builtin_amdgcn_wmma_f32_16x16x32_bf16(
            false, Ah, false, Bh, (short)0, acc[nt], false, false);
      }
    }
  };

  build(0, 0);
  __syncthreads();
#pragma unroll 1
  for (int q = 0; q < 36; ++q) {
    int buf = q & 1;
    if (q + 1 < 36) build(q + 1, buf ^ 1);  // overlap next chunk's gathers
    consume(q, buf);
    __syncthreads();
  }

  // C/D layout: VGPR r holds M=r (lanes 0-15) / M=r+8 (lanes 16-31), N=lane%16.
  float* op = out + ((size_t)(b * COUT + o0) * HH + h) * WW;
#pragma unroll
  for (int nt = 0; nt < 4; ++nt) {
#pragma unroll
    for (int r = 0; r < 8; ++r) {
      op[(size_t)(r + lhi * 8) * HW + nt * 16 + lrow] = acc[nt][r];
    }
  }
}

// ---------------------------------------------------------------------------
extern "C" void kernel_launch(void* const* d_in, const int* in_sizes, int n_in,
                              void* d_out, int out_size, void* d_ws, size_t ws_size,
                              hipStream_t stream) {
  const float* x      = (const float*)d_in[0];
  const float* offset = (const float*)d_in[1];
  const float* weight = (const float*)d_in[2];
  float* out = (float*)d_out;
  unsigned short* wtH = (unsigned short*)d_ws;       // NWT bf16
  unsigned short* wtL = wtH + NWT;                   // NWT bf16 (2.36 MB total)

  pack_weight_kernel<<<(NWT + 255) / 256, 256, 0, stream>>>(weight, wtH, wtL);
  deform_conv_wmma_kernel<<<4 * HH, 512, 0, stream>>>(x, offset, wtH, wtL, out);
}
